// HeteroEdgeBias_52639119179996
// MI455X (gfx1250) — matrix operand order repaired and checked by reading, not compiled, not run on hardware
//
#include <hip/hip_runtime.h>
#include <hip/hip_bf16.h>

typedef __attribute__((ext_vector_type(16))) _Float16 v16h;
typedef __attribute__((ext_vector_type(8)))  float    v8f;
typedef __attribute__((ext_vector_type(4)))  float    f4;

// Problem constants (fixed by the reference file)
static constexpr int Bc = 16;   // batch
static constexpr int Hc = 16;   // heads
static constexpr int Sc = 512;  // max_seq_len
static constexpr int Tc = 16;   // num_edge_types (embedding has T+1 = 17 rows)

// ---------------------------------------------------------------------------
// Kernel 1: stream 256 MiB of zeros with non-temporal 128-bit stores.
// Output (256 MiB) > L2 (192 MB) -> NT is the right temporal hint.
// This is the roofline term: ~11.5 us at 23.3 TB/s.
// ---------------------------------------------------------------------------
__global__ void HeteroEdgeBias_zero_kernel(f4* __restrict__ out4, long n4,
                                           float* __restrict__ out, long n) {
    const long stride = (long)gridDim.x * blockDim.x;
    const long tid = (long)blockIdx.x * blockDim.x + threadIdx.x;
    f4 z = {0.0f, 0.0f, 0.0f, 0.0f};
    for (long i = tid; i < n4; i += stride) {
        __builtin_nontemporal_store(z, out4 + i);
    }
    if (n4 * 4 < n) {  // scalar tail (never taken for 2^26 elements; stay general)
        for (long j = n4 * 4 + tid; j < n; j += stride) {
            __builtin_nontemporal_store(0.0f, out + j);
        }
    }
}

// ---------------------------------------------------------------------------
// Kernel 2: one wave32 handles 16 edges.
//   vals[16 edges][16 heads] = onehot(edge_type) x edge_embedding
// via two accumulating V_WMMA_F32_16X16X32_F16 (embedding split into f16 hi +
// f16 residual so the gather is exact to ~1e-7, accumulated in f32).
// Each lane precomputes the 26-bit output offset of "its" edge once; the 8
// scatter stores pick up the offset they need with ds_bpermute_b32 instead of
// re-walking a 4-deep dependent global-load chain per store.
// ---------------------------------------------------------------------------
__global__ void HeteroEdgeBias_scatter_kernel(
        const int*   __restrict__ edge_index,  // [2, E]
        const int*   __restrict__ edge_type,   // [E]
        const int*   __restrict__ batch_vec,   // [B*S]
        const int*   __restrict__ offsets,     // [B]
        const float* __restrict__ emb,         // [(T+1), H] row-major
        float*       __restrict__ out,         // [B, H, S, S]
        int E) {
    const int lane = threadIdx.x & 31;
    const int nGroups = (E + 15) >> 4;
    const int g = blockIdx.x * 8 + (threadIdx.x >> 5); // 8 waves / 256-thr block
    if (g >= nGroups) return;                          // wave-uniform: EXEC stays full

    const int base = g << 4;                           // first edge of this group
    const int n  = lane & 15;                          // head / column / row index
    const int hi = lane >> 4;                          // lane half (0 or 1)

    // ---- per-lane edge -> output offset (branchless; clamped for ragged tail)
    const int e0 = base + n;
    const int eC = (e0 < E) ? e0 : (E - 1);
    const int src = edge_index[eC];
    const int dst = edge_index[E + eC];
    const int b   = batch_vec[src];
    const int off = offsets[b];
    // p = b*H*S*S + s*S + d  (< 2^26, fits int32)
    const int p = b * (Hc * Sc * Sc) + (src - off) * Sc + (dst - off);

    // ---- B matrix (32x16 f16): lane holds column N=n, element i -> K = 16*hi + i
    v16h bh, bl;
#pragma unroll
    for (int i = 0; i < 16; ++i) {
        const int K = (hi << 4) + i;
        const float x = (K <= Tc) ? emb[K * Hc + n] : 0.0f;
        const _Float16 xh = (_Float16)x;
        const _Float16 xl = (_Float16)(x - (float)xh);
        bh[i] = xh;
        bl[i] = xl;
    }

    // ---- A matrix (16x32 f16 onehot): lane holds row M=n.
    // element i -> K = 16*hi + i
    const int t = (e0 < E) ? edge_type[eC] : -1;
    v16h a;
#pragma unroll
    for (int i = 0; i < 16; ++i) {
        const int K = (hi << 4) + i;
        a[i] = (K == t) ? (_Float16)1.0f : (_Float16)0.0f;
    }

    // ---- D = A*B_lo + A*B_hi  (f32 accumulate)
    v8f c = {};
    c = __builtin_amdgcn_wmma_f32_16x16x32_f16(false, a, false, bl,
                                               (short)0, c, false, false);
    c = __builtin_amdgcn_wmma_f32_16x16x32_f16(false, a, false, bh,
                                               (short)0, c, false, false);

    // ---- scatter: VGPR r of lane holds (edge m = r + 8*hi, head n)
    const int hbase = n * (Sc * Sc);                   // head offset within batch
    if (base + 16 <= E) {                              // uniform fast path (always, E%16==0)
#pragma unroll
        for (int r = 0; r < 8; ++r) {
            const int ps = __builtin_amdgcn_ds_bpermute((r + (hi << 3)) << 2, p);
            __builtin_nontemporal_store(c[r], out + (long)(unsigned)(ps + hbase));
        }
    } else {                                           // ragged tail group
#pragma unroll
        for (int r = 0; r < 8; ++r) {
            const int ps = __builtin_amdgcn_ds_bpermute((r + (hi << 3)) << 2, p);
            if (base + r + (hi << 3) < E) {
                __builtin_nontemporal_store(c[r], out + (long)(unsigned)(ps + hbase));
            }
        }
    }
}

extern "C" void kernel_launch(void* const* d_in, const int* in_sizes, int n_in,
                              void* d_out, int out_size, void* d_ws, size_t ws_size,
                              hipStream_t stream) {
    const int*   edge_index = (const int*)d_in[0];   // [2, E]
    const int*   edge_type  = (const int*)d_in[1];   // [E]
    const int*   batch_vec  = (const int*)d_in[2];   // [B*S]
    // d_in[3] = batch_size (scalar), d_in[4] = max_seq_len (scalar): compile-time here
    const int*   offsets    = (const int*)d_in[5];   // [B]
    const float* emb        = (const float*)d_in[6]; // [(T+1)*H]
    float*       out        = (float*)d_out;

    const int  E  = in_sizes[1];
    const long nO = (long)out_size;       // B*H*S*S = 2^26 elements (256 MiB f32)
    const long n4 = nO >> 2;

    // 1) zero-fill 256 MiB with NT b128 stores (bandwidth-bound bulk)
    HeteroEdgeBias_zero_kernel<<<8192, 256, 0, stream>>>((f4*)out, n4, out, nO);

    // 2) WMMA gather + bpermute-fed scatter (one wave per 16 edges)
    const int nGroups = (E + 15) >> 4;
    const int blocks  = (nGroups + 7) >> 3;           // 8 wave32 per block
    HeteroEdgeBias_scatter_kernel<<<blocks, 256, 0, stream>>>(
        edge_index, edge_type, batch_vec, offsets, emb, out, E);
}
